// xDeepFMExtractor_41575283425692
// MI455X (gfx1250) — matrix operand order repaired and checked by reading, not compiled, not run on hardware
//
#include <hip/hip_runtime.h>

// ---------------------------------------------------------------------------
// Types / helpers
// ---------------------------------------------------------------------------
typedef __attribute__((ext_vector_type(16))) __bf16 v16bf;
typedef __attribute__((ext_vector_type(8)))  __bf16 v8bf;
typedef __attribute__((ext_vector_type(8)))  float  v8f;

union Frag16 { uint4 u[2]; v16bf v; };
union RunU   { uint4 u; v8bf v; };

__device__ __forceinline__ unsigned f2bf(float f) {
    unsigned u = __float_as_uint(f);
    return (u + 0x7fffu + ((u >> 16) & 1u)) >> 16;   // RNE to bf16
}
__device__ __forceinline__ v8f wmma_bf16(v16bf a, v16bf b, v8f c) {
    return __builtin_amdgcn_wmma_f32_16x16x32_bf16(false, a, false, b, (short)0, c, false, false);
}

// ---------------------------------------------------------------------------
// Weight prepack: fp32 [O][srcK] row-major  ->  bf16 B-fragment-major
// dst[((step*CT+ct)*32 + lane)*16 + e] = W[o = ct*16 + (lane&15)]
//                                         [kp = step*32 + (lane>=16?8:0) + (e&7) + (e>=8?16:0)]
// logical kp decomposes as m*nfPad + n ; entries with n >= nfReal are zero pad.
// ---------------------------------------------------------------------------
__global__ void pack_w_kernel(const float* __restrict__ src,
                              unsigned short* __restrict__ dst,
                              int ctn, int nfPad, int nfReal, int srcK, int total) {
    int gid = blockIdx.x * 256 + threadIdx.x;
    if (gid >= total) return;
    int e    = gid & 15;
    int lane = (gid >> 4) & 31;
    int rem  = gid >> 9;
    int c    = rem % ctn;
    int step = rem / ctn;
    int o    = c * 16 + (lane & 15);
    int kp   = step * 32 + ((lane >> 4) << 3) + ((e >> 3) << 4) + (e & 7);
    int m    = kp / nfPad;
    int n    = kp - m * nfPad;
    float v  = (n < nfReal) ? src[(size_t)o * srcK + (size_t)m * nfReal + n] : 0.0f;
    dst[gid] = (unsigned short)f2bf(v);
}

__global__ void cvt_bf16_kernel(const float* __restrict__ src,
                                unsigned short* __restrict__ dst, int n) {
    int i = blockIdx.x * 256 + threadIdx.x;
    if (i < n) dst[i] = (unsigned short)f2bf(src[i]);
}

// ---------------------------------------------------------------------------
// CIN layer as WMMA GEMM with on-the-fly outer-product A formation.
//   rows r = (b, d)  (B*32 of them), K = 40*NF (NF=48 padded / 64), N = 128.
// One wave per batch b: 2 row-tiles (d=0..15, 16..31) x 8 col-tiles.
// A-frag runs are built with packed bf16 multiplies (v_pk_mul_bf16).
// Layer1 (!L2): factors are x0 with itself (NF=48, n>=40 zero-padded);
//   cols 0..63 -> h1[b][d][n] bf16 ; cols 64..127 -> sum_d relu -> cinf[b][0..63]
// Layer2 (L2): factors x0 (m) with h1 (n, NF=64);
//   all cols -> sum_d relu -> cinf[b][64..191]
// ---------------------------------------------------------------------------
template <int NF, int STEPS, bool L2>
__global__ __launch_bounds__(256) void cin_kernel(
    const float* __restrict__ x0,            // [B,40,32] fp32
    const unsigned short* __restrict__ h1,   // [B,32,64] bf16 (L2 only)
    const uint4* __restrict__ Wp,            // prepacked B frags
    const float* __restrict__ bias,          // [128]
    unsigned short* __restrict__ h1out,      // [B,32,64] bf16 (layer1 only)
    float* __restrict__ cinf)                // [B,192] fp32
{
    extern __shared__ unsigned short smem[];
    const int lane = threadIdx.x & 31;
    const int wave = threadIdx.x >> 5;
    const int b    = blockIdx.x * 8 + wave;

    unsigned short* V0T = smem + wave * (32 * 40 + 32 * NF);  // [d][m] bf16
    unsigned short* VHT = V0T + 32 * 40;                      // [d][n] bf16

    // Stage transposed bf16 factor tiles for this wave's batch (lane == d).
    {
        const float* xb = x0 + (size_t)b * 1280;
        #pragma unroll
        for (int m = 0; m < 40; ++m) {
            unsigned short hv = (unsigned short)f2bf(xb[m * 32 + lane]);
            V0T[lane * 40 + m] = hv;
            if constexpr (!L2) VHT[lane * NF + m] = hv;
        }
        if constexpr (!L2) {
            #pragma unroll
            for (int n = 40; n < NF; ++n) VHT[lane * NF + n] = 0;
        } else {
            const uint4* hp = (const uint4*)(h1 + ((size_t)b * 32 + lane) * 64);
            uint4* vp = (uint4*)(VHT + lane * 64);
            #pragma unroll
            for (int i = 0; i < 8; ++i) vp[i] = hp[i];
        }
    }
    __syncthreads();

    const int li = lane & 15, hi = lane >> 4;
    v8f acc[2][8] = {};

    for (int step = 0; step < STEPS; ++step) {
        const int k0 = step * 32;
        Frag16 A[2];
        #pragma unroll
        for (int rt = 0; rt < 2; ++rt) {
            const int d = rt * 16 + li;
            const __bf16* v0r = (const __bf16*)(V0T + d * 40);
            const unsigned short* vhr = VHT + d * NF;
            #pragma unroll
            for (int rs = 0; rs < 2; ++rs) {
                // 8-aligned K-run: k = k0 + hi*8 + rs*16 + j, j=0..7 (one m)
                const int kk  = k0 + hi * 8 + rs * 16;
                const int m   = kk / NF;
                const int nst = kk - m * NF;
                const __bf16 v0 = v0r[m];
                RunU vh;
                vh.u = *(const uint4*)(vhr + nst);     // 8 bf16 (ds_load_b128)
                RunU pr;
                pr.v = vh.v * v0;                      // 4x v_pk_mul_bf16
                A[rt].u[rs] = pr.u;
            }
        }
        const uint4* bp = Wp + (size_t)step * 512 + lane * 2;
        #pragma unroll
        for (int ct = 0; ct < 8; ++ct) {
            Frag16 Bf;
            Bf.u[0] = bp[ct * 64];
            Bf.u[1] = bp[ct * 64 + 1];
            acc[0][ct] = wmma_bf16(A[0].v, Bf.v, acc[0][ct]);
            acc[1][ct] = wmma_bf16(A[1].v, Bf.v, acc[1][ct]);
        }
    }

    // ---------------- epilogue ----------------
    if constexpr (!L2) {
        // cols 0..63: hidden h1 (bf16, layout [b][d][n])
        #pragma unroll
        for (int ct = 0; ct < 4; ++ct) {
            const int o = ct * 16 + li;
            const float bo = bias[o];
            #pragma unroll
            for (int rt = 0; rt < 2; ++rt) {
                #pragma unroll
                for (int v = 0; v < 8; ++v) {
                    const int d = rt * 16 + hi * 8 + v;
                    float val = fmaxf(acc[rt][ct][v] + bo, 0.0f);
                    h1out[((size_t)b * 32 + d) * 64 + o] = (unsigned short)f2bf(val);
                }
            }
        }
        // cols 64..127: direct features, sum relu over d
        #pragma unroll
        for (int ct = 4; ct < 8; ++ct) {
            const int o = ct * 16 + li;
            const float bo = bias[o];
            float s = 0.0f;
            #pragma unroll
            for (int rt = 0; rt < 2; ++rt)
                #pragma unroll
                for (int v = 0; v < 8; ++v)
                    s += fmaxf(acc[rt][ct][v] + bo, 0.0f);
            s += __shfl_xor(s, 16, 32);   // combine the two lane halves
            if (hi == 0) cinf[(size_t)b * 192 + (ct - 4) * 16 + li] = s;
        }
    } else {
        #pragma unroll
        for (int ct = 0; ct < 8; ++ct) {
            const int o = ct * 16 + li;
            const float bo = bias[o];
            float s = 0.0f;
            #pragma unroll
            for (int rt = 0; rt < 2; ++rt)
                #pragma unroll
                for (int v = 0; v < 8; ++v)
                    s += fmaxf(acc[rt][ct][v] + bo, 0.0f);
            s += __shfl_xor(s, 16, 32);
            if (hi == 0) cinf[(size_t)b * 192 + 64 + ct * 16 + li] = s;
        }
    }
}

// ---------------------------------------------------------------------------
// Deep tower layer: Out = relu(LN(In @ W^T + b)) ; bf16 WMMA, fp32 accum.
// 32 rows per WG; waves 0-3 -> row-tile 0, waves 4-7 -> row-tile 1;
// each wave owns CTW = N/64 col-tiles. Deterministic LN via LDS partials.
// ---------------------------------------------------------------------------
template <int K, int N, bool F32OUT>
__global__ __launch_bounds__(256) void deep_kernel(
    const unsigned short* __restrict__ in,   // [B,K] bf16
    const uint4* __restrict__ Wp,            // prepacked B frags
    const float* __restrict__ bias,
    const float* __restrict__ gamma,
    const float* __restrict__ beta,
    void* __restrict__ outv)                 // bf16 [B,N] or fp32 [B,N]
{
    constexpr int CT    = N / 16;
    constexpr int CTW   = CT / 4;
    constexpr int STEPS = K / 32;
    __shared__ float pS[32][4];
    __shared__ float pS2[32][4];
    __shared__ float muA[32];
    __shared__ float rsA[32];

    const int lane = threadIdx.x & 31;
    const int wave = threadIdx.x >> 5;
    const int rt   = wave >> 2;
    const int cg   = wave & 3;
    const int li   = lane & 15, hi = lane >> 4;
    const size_t rowBase = (size_t)blockIdx.x * 32;
    const unsigned short* rp = in + (rowBase + rt * 16 + li) * K;

    v8f acc[CTW] = {};
    for (int step = 0; step < STEPS; ++step) {
        const uint4* ap = (const uint4*)(rp + step * 32 + hi * 8);
        Frag16 A;
        A.u[0] = ap[0];   // K-run k0+hi*8 .. +7
        A.u[1] = ap[2];   // K-run +16
        const uint4* bp = Wp + (size_t)step * (CT * 64) + lane * 2;
        #pragma unroll
        for (int j = 0; j < CTW; ++j) {
            const int ct = cg * CTW + j;
            Frag16 Bf;
            Bf.u[0] = bp[ct * 64];
            Bf.u[1] = bp[ct * 64 + 1];
            acc[j] = wmma_bf16(A.v, Bf.v, acc[j]);
        }
    }

    // bias
    #pragma unroll
    for (int j = 0; j < CTW; ++j) {
        const int col = (cg * CTW + j) * 16 + li;
        const float bc = bias[col];
        #pragma unroll
        for (int v = 0; v < 8; ++v) acc[j][v] += bc;
    }

    // per-row partial sums for LN (within-wave shuffle reduce, width 16)
    #pragma unroll
    for (int v = 0; v < 8; ++v) {
        float s = 0.0f, s2 = 0.0f;
        #pragma unroll
        for (int j = 0; j < CTW; ++j) { float t = acc[j][v]; s += t; s2 += t * t; }
        #pragma unroll
        for (int m = 8; m >= 1; m >>= 1) {
            s  += __shfl_xor(s,  m, 16);
            s2 += __shfl_xor(s2, m, 16);
        }
        if (li == 0) {
            const int row = rt * 16 + hi * 8 + v;
            pS[row][cg] = s;
            pS2[row][cg] = s2;
        }
    }
    __syncthreads();
    if (threadIdx.x < 32) {
        const int r = threadIdx.x;
        float s  = pS[r][0] + pS[r][1] + pS[r][2] + pS[r][3];
        float s2 = pS2[r][0] + pS2[r][1] + pS2[r][2] + pS2[r][3];
        float mu  = s / (float)N;
        float var = s2 / (float)N - mu * mu;
        muA[r] = mu;
        rsA[r] = rsqrtf(var + 1e-5f);
    }
    __syncthreads();

    #pragma unroll
    for (int j = 0; j < CTW; ++j) {
        const int col = (cg * CTW + j) * 16 + li;
        const float g = gamma[col], be = beta[col];
        #pragma unroll
        for (int v = 0; v < 8; ++v) {
            const int row = rt * 16 + hi * 8 + v;
            float y = (acc[j][v] - muA[row]) * rsA[row] * g + be;
            y = fmaxf(y, 0.0f);
            const size_t oi = (rowBase + row) * N + col;
            if constexpr (F32OUT) ((float*)outv)[oi] = y;
            else ((unsigned short*)outv)[oi] = (unsigned short)f2bf(y);
        }
    }
}

// ---------------------------------------------------------------------------
// aux[b] = x[b]·lin_w + lin_b + cin_feat[b]·cin_out_w + cin_out_b
// one wave per row (memory bound)
// ---------------------------------------------------------------------------
__global__ __launch_bounds__(256) void aux_kernel(
    const float* __restrict__ x, const float* __restrict__ cinf,
    const float* __restrict__ lin_w, const float* __restrict__ lin_b,
    const float* __restrict__ cow, const float* __restrict__ cob,
    float* __restrict__ aux)
{
    const int lane = threadIdx.x & 31;
    const int wave = threadIdx.x >> 5;
    const int row  = blockIdx.x * 8 + wave;
    const float* xr = x + (size_t)row * 1280;
    float s = 0.0f;
    for (int k = lane; k < 1280; k += 32) s += xr[k] * lin_w[k];
    const float* cr = cinf + (size_t)row * 192;
    for (int k = lane; k < 192; k += 32) s += cr[k] * cow[k];
    #pragma unroll
    for (int m = 16; m >= 1; m >>= 1) s += __shfl_xor(s, m, 32);
    if (lane == 0) aux[row] = s + lin_b[0] + cob[0];
}

// ---------------------------------------------------------------------------
// Host launcher
// ---------------------------------------------------------------------------
extern "C" void kernel_launch(void* const* d_in, const int* in_sizes, int n_in,
                              void* d_out, int out_size, void* d_ws, size_t ws_size,
                              hipStream_t stream) {
    (void)in_sizes; (void)n_in; (void)out_size; (void)ws_size;
    constexpr int B = 16384;

    const float* x     = (const float*)d_in[0];
    const float* emb   = (const float*)d_in[1];
    const float* lin_w = (const float*)d_in[2];
    const float* lin_b = (const float*)d_in[3];
    const float* cw0   = (const float*)d_in[4];
    const float* cb0   = (const float*)d_in[5];
    const float* cw1   = (const float*)d_in[6];
    const float* cb1   = (const float*)d_in[7];
    const float* cow   = (const float*)d_in[8];
    const float* cob   = (const float*)d_in[9];
    const float* dw0   = (const float*)d_in[10];
    const float* db0   = (const float*)d_in[11];
    const float* g0    = (const float*)d_in[12];
    const float* be0   = (const float*)d_in[13];
    const float* dw1   = (const float*)d_in[14];
    const float* db1   = (const float*)d_in[15];
    const float* g1    = (const float*)d_in[16];
    const float* be1   = (const float*)d_in[17];
    const float* dw2   = (const float*)d_in[18];
    const float* db2   = (const float*)d_in[19];
    const float* g2    = (const float*)d_in[20];
    const float* be2   = (const float*)d_in[21];

    char* ws = (char*)d_ws;
    size_t off = 0;
    auto alloc = [&](size_t bytes) -> char* {
        char* p = ws + off;
        off += (bytes + 255) & ~(size_t)255;
        return p;
    };
    unsigned short* Wp0  = (unsigned short*)alloc((size_t)60 * 8 * 512 * 2);   // CIN1 B frags
    unsigned short* Wp1  = (unsigned short*)alloc((size_t)80 * 8 * 512 * 2);   // CIN2 B frags
    unsigned short* Dp0  = (unsigned short*)alloc((size_t)40 * 16 * 512 * 2);  // deep0
    unsigned short* Dp1  = (unsigned short*)alloc((size_t)8 * 8 * 512 * 2);    // deep1
    unsigned short* Dp2  = (unsigned short*)alloc((size_t)4 * 4 * 512 * 2);    // deep2
    unsigned short* xb   = (unsigned short*)alloc((size_t)B * 1280 * 2);       // bf16 x
    unsigned short* h1b  = (unsigned short*)alloc((size_t)B * 32 * 64 * 2);    // CIN hidden
    float*          cinf = (float*)alloc((size_t)B * 192 * 4);                 // cin features
    unsigned short* hd1  = (unsigned short*)alloc((size_t)B * 256 * 2);        // deep act 1
    unsigned short* hd2  = (unsigned short*)alloc((size_t)B * 128 * 2);        // deep act 2

    // ---- weight prepack + x conversion ----
    pack_w_kernel<<<(60 * 8 * 512 + 255) / 256, 256, 0, stream>>>(cw0, Wp0, 8, 48, 40, 1600, 60 * 8 * 512);
    pack_w_kernel<<<(80 * 8 * 512 + 255) / 256, 256, 0, stream>>>(cw1, Wp1, 8, 64, 64, 2560, 80 * 8 * 512);
    pack_w_kernel<<<(40 * 16 * 512 + 255) / 256, 256, 0, stream>>>(dw0, Dp0, 16, 1280, 1280, 1280, 40 * 16 * 512);
    pack_w_kernel<<<(8 * 8 * 512 + 255) / 256, 256, 0, stream>>>(dw1, Dp1, 8, 256, 256, 256, 8 * 8 * 512);
    pack_w_kernel<<<(4 * 4 * 512 + 255) / 256, 256, 0, stream>>>(dw2, Dp2, 4, 128, 128, 128, 4 * 4 * 512);
    cvt_bf16_kernel<<<(B * 1280 + 255) / 256, 256, 0, stream>>>(x, xb, B * 1280);

    // ---- CIN (the FLOP-dominant part) ----
    const size_t sh1 = 8 * (size_t)(32 * 40 + 32 * 48) * 2;   // 45056 B
    const size_t sh2 = 8 * (size_t)(32 * 40 + 32 * 64) * 2;   // 53248 B
    cin_kernel<48, 60, false><<<B / 8, 256, sh1, stream>>>(emb, nullptr, (const uint4*)Wp0, cb0, h1b, cinf);
    cin_kernel<64, 80, true ><<<B / 8, 256, sh2, stream>>>(emb, h1b, (const uint4*)Wp1, cb1, nullptr, cinf);

    // ---- aux (linear + cin head) ----
    aux_kernel<<<B / 8, 256, 0, stream>>>(x, cinf, lin_w, lin_b, cow, cob,
                                          (float*)d_out + (size_t)B * 64);

    // ---- deep tower ----
    deep_kernel<1280, 256, false><<<B / 32, 256, 0, stream>>>(xb,  (const uint4*)Dp0, db0, g0, be0, hd1);
    deep_kernel<256,  128, false><<<B / 32, 256, 0, stream>>>(hd1, (const uint4*)Dp1, db1, g1, be1, hd2);
    deep_kernel<128,  64,  true ><<<B / 32, 256, 0, stream>>>(hd2, (const uint4*)Dp2, db2, g2, be2, d_out);
}
